// GriffinBlock_80642305950381
// MI455X (gfx1250) — matrix operand order repaired and checked by reading, not compiled, not run on hardware
//
#include <hip/hip_runtime.h>
#include <hip/hip_bf16.h>

// Griffin-style block for MI455X (gfx1250, wave32, WMMA bf16).
// GEMMs: block = 16-row x 128-col strip, 8 waves = 8 n-tiles. The 16x128 bf16
// A-slab is staged into LDS via global_load_async_to_lds_b128 (ASYNCcnt,
// double-buffered) and shared by all 8 waves; B streams from the L2-resident
// bf16 weights. v_wmma_f32_16x16x32_bf16 with fused epilogues everywhere.

#define DD    1024
#define DRR   1024
#define FFN_  4096
#define BB    2
#define TT    4096
#define MTOT  (BB * TT)   // 8192 tokens
#define KSTG  128         // K elements staged per LDS slab

typedef __attribute__((ext_vector_type(16))) __bf16 v16bf;
typedef __attribute__((ext_vector_type(8)))  float  v8f;

union FragAB { v16bf v; uint4 q[2]; };

__device__ __forceinline__ unsigned short f2bf(float x) {
  unsigned int u = __float_as_uint(x);
  unsigned int r = ((u >> 16) & 1u) + 0x7FFFu;   // round-to-nearest-even
  return (unsigned short)((u + r) >> 16);
}
__device__ __forceinline__ float bf2f(unsigned short h) {
  return __uint_as_float(((unsigned int)h) << 16);
}
__device__ __forceinline__ float sigmoidf_(float x) {
  return 1.0f / (1.0f + __expf(-x));
}

// B-matrix 32x16 bf16 fragment (B stored transposed: row = output column n,
// contiguous K). Lanes 0-15 hold K {0..7,16..23}, lanes 16-31 K {8..15,24..31}.
__device__ __forceinline__ v16bf load_fragB(const unsigned short* __restrict__ base,
                                            int row, int ld, int k0, int lane) {
  FragAB f;
  const unsigned short* p = base + (size_t)row * ld + k0 + ((lane >> 4) << 3);
  f.q[0] = *(const uint4*)(p);
  f.q[1] = *(const uint4*)(p + 16);
  return f.v;
}

// A-matrix 16x32 bf16 fragment from the LDS slab (ld = KSTG). Per ISA 7.12.2:
// lanes 0-15 hold M=lane, K chunks {0..7,16..23}; lanes 16-31 K {8..15,24..31}.
__device__ __forceinline__ v16bf frag_from_lds(const unsigned short* sm,
                                               int kloc, int lane) {
  FragAB f;
  const unsigned short* p = sm + (lane & 15) * KSTG + kloc + ((lane >> 4) << 3);
  f.q[0] = *(const uint4*)(p);
  f.q[1] = *(const uint4*)(p + 16);
  return f.v;
}

// Async stage of one 16xKSTG bf16 A-slab into LDS. Each of the 256 threads
// carries one 16B chunk: 8 waves x 32 lanes x 16B = 4KB. GVS addressing:
// saddr = matrix base, vaddr = 32-bit byte offset, vdst = LDS byte address.
__device__ __forceinline__ void issue_tileA(const unsigned short* Abase, int ldA,
                                            int m0, int k, unsigned smbase,
                                            int buf, int tid) {
  const int crow = tid >> 4;            // 0..15
  const int cchk = (tid & 15) << 3;     // element offset 0,8,...,120
  unsigned loff = smbase + (unsigned)((buf * 16 * KSTG + crow * KSTG + cchk) * 2);
  unsigned goff = (unsigned)(((size_t)(m0 + crow) * (size_t)ldA + (size_t)(k + cchk)) * 2);
  asm volatile("global_load_async_to_lds_b128 %0, %1, %2"
               :: "v"(loff), "v"(goff), "s"(Abase) : "memory");
}
__device__ __forceinline__ void wait_async_le1() {
  asm volatile("s_wait_asynccnt 0x1" ::: "memory");
}
__device__ __forceinline__ void wait_async_0() {
  asm volatile("s_wait_asynccnt 0x0" ::: "memory");
}

// ---------------------------------------------------------------- utilities
__global__ void __launch_bounds__(256)
k_f32_to_bf16(const float* __restrict__ src, unsigned short* __restrict__ dst, int n) {
  int i = blockIdx.x * 256 + threadIdx.x;
  if (i < n) dst[i] = f2bf(src[i]);
}

// RMSNorm over D=1024, one block per token, bf16 output.
__global__ void __launch_bounds__(256)
k_rmsnorm_bf16(const float* __restrict__ x, const float* __restrict__ w,
               unsigned short* __restrict__ out) {
  __shared__ float red[256];
  const int m = blockIdx.x;
  const float* xr = x + (size_t)m * DD;
  float ss = 0.0f;
  for (int i = threadIdx.x; i < DD; i += 256) { float v = xr[i]; ss += v * v; }
  red[threadIdx.x] = ss;
  __syncthreads();
  for (int s = 128; s > 0; s >>= 1) {
    if (threadIdx.x < s) red[threadIdx.x] += red[threadIdx.x + s];
    __syncthreads();
  }
  const float scale = rsqrtf(red[0] * (1.0f / DD) + 1e-6f);
  for (int i = threadIdx.x; i < DD; i += 256)
    out[(size_t)m * DD + i] = f2bf(xr[i] * scale * w[i]);
}

// Depthwise causal conv, K=4, bf16 in/out, f32 taps/accum.
__global__ void __launch_bounds__(256)
k_conv(const unsigned short* __restrict__ hn, const float* __restrict__ cw,
       const float* __restrict__ cb, unsigned short* __restrict__ h) {
  const int idx = blockIdx.x * 256 + threadIdx.x;     // over MTOT*DD
  const int d = idx & (DD - 1);
  const int m = idx / DD;
  const int t = m & (TT - 1);
  float acc = cb[d];
#pragma unroll
  for (int kk = 0; kk < 4; ++kk) {
    const int tt = t - 3 + kk;
    if (tt >= 0) acc += bf2f(hn[(size_t)(m - 3 + kk) * DD + d]) * cw[d * 4 + kk];
  }
  h[idx] = f2bf(acc);
}

// Per-channel linear recurrence s_t = a_t*s_{t-1} + v_t, bf16 output.
__global__ void __launch_bounds__(256)
k_scan(const float* __restrict__ a, const float* __restrict__ sig,
       unsigned short* __restrict__ rec) {
  const int c = blockIdx.x * 256 + threadIdx.x;       // B*DRR channels
  const int b = c / DRR, d = c % DRR;
  const size_t base = (size_t)b * TT * DRR + d;
  float s = 0.0f;
  for (int t = 0; t < TT; ++t) {
    const size_t i = base + (size_t)t * DRR;
    s = a[i] * s + sig[i];
    rec[i] = f2bf(s);
  }
}

// ----------------------------------------------- fused a/i/v projection GEMM
__global__ void __launch_bounds__(256)
k_gemm_aiv(const unsigned short* __restrict__ hbf,
           const unsigned short* __restrict__ wa,
           const unsigned short* __restrict__ wi,
           const unsigned short* __restrict__ wv,
           const float* __restrict__ decay_bias,
           float* __restrict__ a_out, float* __restrict__ sig_out) {
  __shared__ unsigned short smA[2][16][KSTG];
  const int tid  = threadIdx.x;
  const int lane = tid & 31;
  const int wave = tid >> 5;
  const int ngrp = DRR / 128;
  const int m0 = (blockIdx.x / ngrp) * 16;
  const int n0 = (blockIdx.x % ngrp) * 128 + wave * 16;
  const int brow = n0 + (lane & 15);
  const unsigned smbase = (unsigned)(size_t)&smA[0][0][0];
  v8f ca = {}, ci = {}, cv = {};
  issue_tileA(hbf, DD, m0, 0, smbase, 0, tid);
  const int stages = DD / KSTG;
  for (int s = 0; s < stages; ++s) {
    const int k = s * KSTG;
    if (s + 1 < stages) { issue_tileA(hbf, DD, m0, k + KSTG, smbase, (s + 1) & 1, tid); wait_async_le1(); }
    else                { wait_async_0(); }
    __syncthreads();
    const unsigned short* As = &smA[s & 1][0][0];
#pragma unroll
    for (int kl = 0; kl < KSTG; kl += 32) {
      __builtin_prefetch(wa + (size_t)brow * DD + k + kl + 64, 0, 1);
      v16bf af  = frag_from_lds(As, kl, lane);
      v16bf bfa = load_fragB(wa, brow, DD, k + kl, lane);
      v16bf bfi = load_fragB(wi, brow, DD, k + kl, lane);
      v16bf bfv = load_fragB(wv, brow, DD, k + kl, lane);
      ca = __builtin_amdgcn_wmma_f32_16x16x32_bf16(false, af, false, bfa, (short)0, ca, false, false);
      ci = __builtin_amdgcn_wmma_f32_16x16x32_bf16(false, af, false, bfi, (short)0, ci, false, false);
      cv = __builtin_amdgcn_wmma_f32_16x16x32_bf16(false, af, false, bfv, (short)0, cv, false, false);
    }
    __syncthreads();
  }
  const int n = n0 + (lane & 15);
  const float db = decay_bias[n];
  const int mrow0 = m0 + ((lane >> 4) << 3);
#pragma unroll
  for (int r = 0; r < 8; ++r) {
    const size_t idx = (size_t)(mrow0 + r) * DRR + n;
    const float aval = sigmoidf_(ca[r] + db);
    const float ival = sigmoidf_(ci[r]);
    a_out[idx]   = aval;
    sig_out[idx] = sqrtf(fmaxf(1.0f - aval * aval, 1e-8f)) * ival * cv[r];
  }
}

// Generic WMMA GEMM with residual-add epilogue: out = resid + A @ B^T.
__global__ void __launch_bounds__(256)
k_gemm_resid(const unsigned short* __restrict__ A,
             const unsigned short* __restrict__ Bm,
             const float* __restrict__ resid, float* __restrict__ out,
             int N, int K) {
  __shared__ unsigned short smA[2][16][KSTG];
  const int tid  = threadIdx.x;
  const int lane = tid & 31;
  const int wave = tid >> 5;
  const int ngrp = N / 128;
  const int m0 = (blockIdx.x / ngrp) * 16;
  const int n0 = (blockIdx.x % ngrp) * 128 + wave * 16;
  const int brow = n0 + (lane & 15);
  const unsigned smbase = (unsigned)(size_t)&smA[0][0][0];
  v8f c = {};
  issue_tileA(A, K, m0, 0, smbase, 0, tid);
  const int stages = K / KSTG;
  for (int s = 0; s < stages; ++s) {
    const int k = s * KSTG;
    if (s + 1 < stages) { issue_tileA(A, K, m0, k + KSTG, smbase, (s + 1) & 1, tid); wait_async_le1(); }
    else                { wait_async_0(); }
    __syncthreads();
    const unsigned short* As = &smA[s & 1][0][0];
#pragma unroll
    for (int kl = 0; kl < KSTG; kl += 32) {
      __builtin_prefetch(Bm + (size_t)brow * K + k + kl + 64, 0, 1);
      v16bf af = frag_from_lds(As, kl, lane);
      v16bf bf = load_fragB(Bm, brow, K, k + kl, lane);
      c = __builtin_amdgcn_wmma_f32_16x16x32_bf16(false, af, false, bf, (short)0, c, false, false);
    }
    __syncthreads();
  }
  const int n = n0 + (lane & 15);
  const int mrow0 = m0 + ((lane >> 4) << 3);
#pragma unroll
  for (int r = 0; r < 8; ++r) {
    const size_t idx = (size_t)(mrow0 + r) * N + n;
    out[idx] = resid[idx] + c[r];
  }
}

// Fused gate/up GEMM: both tiles share A fragments; SwiGLU epilogue -> bf16.
__global__ void __launch_bounds__(256)
k_gemm_glu(const unsigned short* __restrict__ hn,
           const unsigned short* __restrict__ wgu,
           unsigned short* __restrict__ act) {
  __shared__ unsigned short smA[2][16][KSTG];
  const int tid  = threadIdx.x;
  const int lane = tid & 31;
  const int wave = tid >> 5;
  const int ngrp = FFN_ / 128;
  const int m0 = (blockIdx.x / ngrp) * 16;
  const int n0 = (blockIdx.x % ngrp) * 128 + wave * 16;
  const int grow = n0 + (lane & 15);           // gate rows [0, FFN)
  const int urow = grow + FFN_;                // up rows   [FFN, 2FFN)
  const unsigned smbase = (unsigned)(size_t)&smA[0][0][0];
  v8f cg = {}, cu = {};
  issue_tileA(hn, DD, m0, 0, smbase, 0, tid);
  const int stages = DD / KSTG;
  for (int s = 0; s < stages; ++s) {
    const int k = s * KSTG;
    if (s + 1 < stages) { issue_tileA(hn, DD, m0, k + KSTG, smbase, (s + 1) & 1, tid); wait_async_le1(); }
    else                { wait_async_0(); }
    __syncthreads();
    const unsigned short* As = &smA[s & 1][0][0];
#pragma unroll
    for (int kl = 0; kl < KSTG; kl += 32) {
      __builtin_prefetch(wgu + (size_t)grow * DD + k + kl + 64, 0, 1);
      v16bf af = frag_from_lds(As, kl, lane);
      v16bf bg = load_fragB(wgu, grow, DD, k + kl, lane);
      v16bf bu = load_fragB(wgu, urow, DD, k + kl, lane);
      cg = __builtin_amdgcn_wmma_f32_16x16x32_bf16(false, af, false, bg, (short)0, cg, false, false);
      cu = __builtin_amdgcn_wmma_f32_16x16x32_bf16(false, af, false, bu, (short)0, cu, false, false);
    }
    __syncthreads();
  }
  const int n = n0 + (lane & 15);
  const int mrow0 = m0 + ((lane >> 4) << 3);
#pragma unroll
  for (int r = 0; r < 8; ++r) {
    const float g = cg[r];
    act[(size_t)(mrow0 + r) * FFN_ + n] = f2bf(g * sigmoidf_(g) * cu[r]);
  }
}

// ---------------------------------------------------------------------------
extern "C" void kernel_launch(void* const* d_in, const int* in_sizes, int n_in,
                              void* d_out, int out_size, void* d_ws, size_t ws_size,
                              hipStream_t stream) {
  (void)in_sizes; (void)n_in; (void)out_size; (void)ws_size;
  const float* x   = (const float*)d_in[0];
  const float* pnw = (const float*)d_in[1];
  const float* cw  = (const float*)d_in[2];
  const float* cb  = (const float*)d_in[3];
  const float* wa  = (const float*)d_in[4];
  const float* wi  = (const float*)d_in[5];
  const float* wv  = (const float*)d_in[6];
  const float* wo  = (const float*)d_in[7];
  const float* db  = (const float*)d_in[8];
  const float* fnw = (const float*)d_in[9];
  const float* wgu = (const float*)d_in[10];
  const float* wd  = (const float*)d_in[11];

  char* ws = (char*)d_ws;
  const size_t MB = 1024ull * 1024ull;
  unsigned short* wa_bf  = (unsigned short*)(ws + 0 * MB);    //  2 MB
  unsigned short* wi_bf  = (unsigned short*)(ws + 2 * MB);    //  2 MB
  unsigned short* wv_bf  = (unsigned short*)(ws + 4 * MB);    //  2 MB
  unsigned short* wo_bf  = (unsigned short*)(ws + 6 * MB);    //  2 MB
  unsigned short* wgu_bf = (unsigned short*)(ws + 8 * MB);    // 16 MB
  unsigned short* wd_bf  = (unsigned short*)(ws + 24 * MB);   //  8 MB
  unsigned short* h_bf   = (unsigned short*)(ws + 32 * MB);   // 16 MB
  unsigned short* hn_bf  = (unsigned short*)(ws + 48 * MB);   // 16 MB (reused)
  float*          a_f    = (float*)(ws + 64 * MB);            // 32 MB
  float*          sig_f  = (float*)(ws + 96 * MB);            // 32 MB
  unsigned short* rec_bf = (unsigned short*)(ws + 128 * MB);  // 16 MB
  float*          x1_f   = (float*)(ws + 144 * MB);           // 32 MB
  unsigned short* act_bf = (unsigned short*)(ws + 176 * MB);  // 64 MB -> 240 MB total

  // 1) fp32 -> bf16 weights (read once; 32 MB bf16 lives in the 192 MB L2)
  k_f32_to_bf16<<<(DRR * DD + 255) / 256, 256, 0, stream>>>(wa, wa_bf, DRR * DD);
  k_f32_to_bf16<<<(DRR * DD + 255) / 256, 256, 0, stream>>>(wi, wi_bf, DRR * DD);
  k_f32_to_bf16<<<(DRR * DD + 255) / 256, 256, 0, stream>>>(wv, wv_bf, DRR * DD);
  k_f32_to_bf16<<<(DD * DRR + 255) / 256, 256, 0, stream>>>(wo, wo_bf, DD * DRR);
  k_f32_to_bf16<<<(2 * FFN_ * DD + 255) / 256, 256, 0, stream>>>(wgu, wgu_bf, 2 * FFN_ * DD);
  k_f32_to_bf16<<<(DD * FFN_ + 255) / 256, 256, 0, stream>>>(wd, wd_bf, DD * FFN_);

  // 2) pre-norm, 3) causal conv
  k_rmsnorm_bf16<<<MTOT, 256, 0, stream>>>(x, pnw, hn_bf);
  k_conv<<<(MTOT * DD) / 256, 256, 0, stream>>>(hn_bf, cw, cb, h_bf);

  // 4) fused a/i/v projections + gating (triple WMMA, LDS-shared A)
  k_gemm_aiv<<<(MTOT / 16) * (DRR / 128), 256, 0, stream>>>(
      h_bf, wa_bf, wi_bf, wv_bf, db, a_f, sig_f);

  // 5) linear recurrence over T
  k_scan<<<(BB * DRR) / 256, 256, 0, stream>>>(a_f, sig_f, rec_bf);

  // 6) out_proj + residual: x1 = x + rec @ Wo^T
  k_gemm_resid<<<(MTOT / 16) * (DD / 128), 256, 0, stream>>>(
      rec_bf, wo_bf, x, x1_f, DD, DRR);

  // 7) ffn norm
  k_rmsnorm_bf16<<<MTOT, 256, 0, stream>>>(x1_f, fnw, hn_bf);

  // 8) fused gate/up GEMM + SwiGLU
  k_gemm_glu<<<(MTOT / 16) * (FFN_ / 128), 256, 0, stream>>>(hn_bf, wgu_bf, act_bf);

  // 9) down proj + residual -> d_out (f32)
  k_gemm_resid<<<(MTOT / 16) * (DD / 128), 256, 0, stream>>>(
      act_bf, wd_bf, x1_f, (float*)d_out, DD, FFN_);
}
